// MultiHeadSelfAttention_17514876633864
// MI455X (gfx1250) — matrix operand order repaired and checked by reading, not compiled, not run on hardware
//
#include <hip/hip_runtime.h>
#include <stdint.h>
#include <stddef.h>

// ---------------------------------------------------------------------------
// MultiHeadSelfAttention for gfx1250 (MI455X), bf16 WMMA pipeline.
// dims: B=2, N=2048, DIM=1024, H=16, HD=64.  scale = DIM^-0.5 = 1/32.
//
// Pipeline:
//  K0  fp32 -> bf16 (x, w_qkv, w_o)
//  K1  qkv GEMM: Q,K stored row-major in qkb[4096][2048];
//      V stored transposed (d-major) in vtb[b][h][d][token] -- the C-tile
//      register layout makes the transposed store a single b128 per lane.
//  K2  flash attention, one wave per (b,h,16-query block); all WMMA operands
//      loaded as contiguous b128; P re-laid out C->A through 1KB LDS.
//  K3  output projection + bias -> fp32 d_out.
// ---------------------------------------------------------------------------

#define BATCH   2
#define SEQ     2048
#define DIM     1024
#define HEADS   16
#define HEADDIM 64
#define QKVDIM  (3 * DIM)
#define ROWS    (BATCH * SEQ)          // 4096
#define SCALE   0.03125f               // 1024^-0.5

typedef __attribute__((ext_vector_type(16))) __bf16 v16bf;
typedef __attribute__((ext_vector_type(8)))  __bf16 v8bf;
typedef __attribute__((ext_vector_type(8)))  float  v8f;

static __device__ __forceinline__ uint16_t f2bfu(float f) {
    union { float f; uint32_t u; } c; c.f = f;
    uint32_t u = c.u;
    u += 0x7FFFu + ((u >> 16) & 1u);          // round-to-nearest-even
    return (uint16_t)(u >> 16);
}
static __device__ __forceinline__ __bf16 f2bf(float f) {
    uint16_t h = f2bfu(f);
    return __builtin_bit_cast(__bf16, h);
}

static __device__ __forceinline__ v8bf ld8(const uint16_t* p) {
    return *reinterpret_cast<const v8bf*>(p); // 16B -> global_load_b128 / ds_load_b128
}
static __device__ __forceinline__ void st8(uint16_t* p, v8bf v) {
    *reinterpret_cast<v8bf*>(p) = v;          // 16B -> global_store_b128
}

static __device__ __forceinline__ v16bf cat16(v8bf lo, v8bf hi) {
    v16bf r;
#pragma unroll
    for (int i = 0; i < 8; ++i) { r[i] = lo[i]; r[i + 8] = hi[i]; }
    return r;
}

// row-wise reductions across the 16-lane groups of a C tile (lane bit4 = M-half)
static __device__ __forceinline__ float rmax16(float v) {
    v = fmaxf(v, __shfl_xor(v, 1, 32));
    v = fmaxf(v, __shfl_xor(v, 2, 32));
    v = fmaxf(v, __shfl_xor(v, 4, 32));
    v = fmaxf(v, __shfl_xor(v, 8, 32));
    return v;
}
static __device__ __forceinline__ float rsum16(float v) {
    v += __shfl_xor(v, 1, 32);
    v += __shfl_xor(v, 2, 32);
    v += __shfl_xor(v, 4, 32);
    v += __shfl_xor(v, 8, 32);
    return v;
}

// ---------------------------------------------------------------------------
// K0: fp32 -> bf16 conversion
// ---------------------------------------------------------------------------
__global__ void cvt_bf16_kernel(const float* __restrict__ in,
                                uint16_t* __restrict__ out, int n) {
    int i = blockIdx.x * blockDim.x + threadIdx.x;
    if (i < n) out[i] = f2bfu(in[i]);
}

// ---------------------------------------------------------------------------
// Wave-level GEMM core: one wave computes a 16(M) x 64(N) fp32 tile of
// C = A * W^T, with A row-major [M,K] bf16 and W row-major [N,K] bf16
// (so B-matrix column n == W row n, contiguous along K).  K in steps of 32.
// ---------------------------------------------------------------------------
static __device__ __forceinline__ void wave_gemm_16x64(
    const uint16_t* __restrict__ arow,   // A + (m0 + (lane&15)) * Kdim
    const uint16_t* __restrict__ wbase,  // W
    int n0, int Kdim, int lane, v8f c[4]) {
    const int half = lane >> 4;
    const int col  = lane & 15;
    const uint16_t* b0 = wbase + (size_t)(n0 +  0 + col) * Kdim;
    const uint16_t* b1 = wbase + (size_t)(n0 + 16 + col) * Kdim;
    const uint16_t* b2 = wbase + (size_t)(n0 + 32 + col) * Kdim;
    const uint16_t* b3 = wbase + (size_t)(n0 + 48 + col) * Kdim;

#pragma unroll 4
    for (int kk = 0; kk < Kdim; kk += 32) {
        // A tile 16x32: elems 0..15 = K[half*16 .. half*16+15]
        v16bf a = cat16(ld8(arow + kk + half * 16),
                        ld8(arow + kk + half * 16 + 8));
        // B tile 32x16: elems 0..15 = K[half*16 .. half*16+15] of column n
        v16bf vb0 = cat16(ld8(b0 + kk + half * 16), ld8(b0 + kk + half * 16 + 8));
        v16bf vb1 = cat16(ld8(b1 + kk + half * 16), ld8(b1 + kk + half * 16 + 8));
        v16bf vb2 = cat16(ld8(b2 + kk + half * 16), ld8(b2 + kk + half * 16 + 8));
        v16bf vb3 = cat16(ld8(b3 + kk + half * 16), ld8(b3 + kk + half * 16 + 8));
        c[0] = __builtin_amdgcn_wmma_f32_16x16x32_bf16(false, a, false, vb0, (short)0, c[0], false, false);
        c[1] = __builtin_amdgcn_wmma_f32_16x16x32_bf16(false, a, false, vb1, (short)0, c[1], false, false);
        c[2] = __builtin_amdgcn_wmma_f32_16x16x32_bf16(false, a, false, vb2, (short)0, c[2], false, false);
        c[3] = __builtin_amdgcn_wmma_f32_16x16x32_bf16(false, a, false, vb3, (short)0, c[3], false, false);
    }
}

// ---------------------------------------------------------------------------
// K1: qkv = x @ w_qkv^T   [4096 x 3072], one wave per 16x64 tile (12288 waves)
//   feature e in [0,2048):  qkb[token][e]         (Q | K, row-major)
//   feature e in [2048,3072): vtb[(b*1024 + e-2048)][token]  (V, d-major)
// ---------------------------------------------------------------------------
__global__ __launch_bounds__(256)
void qkv_gemm_kernel(const uint16_t* __restrict__ xb,
                     const uint16_t* __restrict__ wb,
                     uint16_t* __restrict__ qkb,
                     uint16_t* __restrict__ vtb) {
    const int tid  = blockIdx.x * blockDim.x + threadIdx.x;
    const int wid  = tid >> 5;
    const int lane = tid & 31;
    const int ntiles = QKVDIM / 64;                 // 48
    const int m0 = (wid / ntiles) * 16;
    const int n0 = (wid % ntiles) * 64;
    const int half = lane >> 4, col = lane & 15;

    v8f c[4] = {v8f{}, v8f{}, v8f{}, v8f{}};
    wave_gemm_16x64(xb + (size_t)(m0 + col) * DIM, wb, n0, DIM, lane, c);

    if (n0 < 2 * DIM) {
        // Q/K third: row-major store
#pragma unroll
        for (int t = 0; t < 4; ++t)
#pragma unroll
            for (int r = 0; r < 8; ++r) {
                const int M = half * 8 + r;
                qkb[(size_t)(m0 + M) * (2 * DIM) + n0 + t * 16 + col] = f2bfu(c[t][r]);
            }
    } else {
        // V third: transposed store -- tokens are contiguous in r, one b128/lane
        const int b      = m0 / SEQ;
        const int token0 = (m0 % SEQ) + half * 8;
#pragma unroll
        for (int t = 0; t < 4; ++t) {
            const int e = n0 - 2 * DIM + t * 16 + col;       // h*64 + d
            v8bf pv;
#pragma unroll
            for (int r = 0; r < 8; ++r) pv[r] = f2bf(c[t][r]);
            st8(vtb + (size_t)(b * DIM + e) * SEQ + token0, pv);
        }
    }
}

// ---------------------------------------------------------------------------
// K2: flash attention.  one wave per (b, h, 16-query block): 4096 waves.
// ---------------------------------------------------------------------------
__global__ __launch_bounds__(256)
void attention_kernel(const uint16_t* __restrict__ qkb,
                      const uint16_t* __restrict__ vtb,
                      uint16_t* __restrict__ attnb) {
    const int tid  = blockIdx.x * blockDim.x + threadIdx.x;
    const int wid  = tid >> 5;
    const int lane = tid & 31;
    const int half = lane >> 4, col = lane & 15;

    const int qblk = wid & 127;
    const int h    = (wid >> 7) & 15;
    const int b    = wid >> 11;
    const int q0   = qblk * 16;

    const uint16_t* qrow  = qkb + (size_t)(b * SEQ + q0 + col) * (2 * DIM) + h * HEADDIM;
    const uint16_t* kbase = qkb + (size_t)(b * SEQ) * (2 * DIM) + DIM + h * HEADDIM;
    const uint16_t* vbase = vtb + (size_t)(b * DIM + h * HEADDIM) * SEQ;

    // Q as two 16x32 A tiles (d = 0..31, 32..63), resident for the whole loop
    const v16bf aq0 = cat16(ld8(qrow +  0 + half * 16), ld8(qrow +  8 + half * 16));
    const v16bf aq1 = cat16(ld8(qrow + 32 + half * 16), ld8(qrow + 40 + half * 16));

    __shared__ uint16_t pshare[8][16 * 32];          // P re-layout buffer, 1KB/wave
    uint16_t* pbuf = pshare[threadIdx.x >> 5];

    v8f o[4] = {v8f{}, v8f{}, v8f{}, v8f{}};
    float mrow[8], lrow[8];
#pragma unroll
    for (int r = 0; r < 8; ++r) { mrow[r] = -1e30f; lrow[r] = 0.0f; }

    for (int j = 0; j < SEQ; j += 32) {
        // ---- S = Q K^T for keys j..j+31 (two 16x16 C tiles) ----
        const uint16_t* kr0 = kbase + (size_t)(j + col) * (2 * DIM);       // keys j..j+15
        const uint16_t* kr1 = kbase + (size_t)(j + 16 + col) * (2 * DIM);  // keys j+16..j+31
        v16bf bk0a = cat16(ld8(kr0 +  0 + half * 16), ld8(kr0 +  8 + half * 16));
        v16bf bk0b = cat16(ld8(kr0 + 32 + half * 16), ld8(kr0 + 40 + half * 16));
        v16bf bk1a = cat16(ld8(kr1 +  0 + half * 16), ld8(kr1 +  8 + half * 16));
        v16bf bk1b = cat16(ld8(kr1 + 32 + half * 16), ld8(kr1 + 40 + half * 16));

        v8f s0 = v8f{}, s1 = v8f{};
        s0 = __builtin_amdgcn_wmma_f32_16x16x32_bf16(false, aq0, false, bk0a, (short)0, s0, false, false);
        s0 = __builtin_amdgcn_wmma_f32_16x16x32_bf16(false, aq1, false, bk0b, (short)0, s0, false, false);
        s1 = __builtin_amdgcn_wmma_f32_16x16x32_bf16(false, aq0, false, bk1a, (short)0, s1, false, false);
        s1 = __builtin_amdgcn_wmma_f32_16x16x32_bf16(false, aq1, false, bk1b, (short)0, s1, false, false);

        // ---- V B-tiles from transposed layout: contiguous along keys.
        //      Issued before softmax so latency hides under exp/shuffles. ----
        v16bf bv[4];
#pragma unroll
        for (int t = 0; t < 4; ++t) {
            const uint16_t* vp = vbase + (size_t)(t * 16 + col) * SEQ + j + half * 16;
            bv[t] = cat16(ld8(vp), ld8(vp + 8));
        }

        // ---- online softmax (row = M = half*8 + r, reductions across 16 lanes) ----
#pragma unroll
        for (int r = 0; r < 8; ++r) {
            const float v0 = s0[r] * SCALE;
            const float v1 = s1[r] * SCALE;
            const float tmax = rmax16(fmaxf(v0, v1));
            const float mnew = fmaxf(mrow[r], tmax);
            const float p0 = __expf(v0 - mnew);
            const float p1 = __expf(v1 - mnew);
            const float rs = rsum16(p0 + p1);
            const float alpha = __expf(mrow[r] - mnew);
            lrow[r] = lrow[r] * alpha + rs;
            mrow[r] = mnew;
            const int M = half * 8 + r;
            pbuf[M * 32 + col]      = f2bfu(p0);
            pbuf[M * 32 + 16 + col] = f2bfu(p1);
            o[0][r] *= alpha; o[1][r] *= alpha; o[2][r] *= alpha; o[3][r] *= alpha;
        }

        // ---- reload P as a 16x32 A tile (same wave: DS ops are in-order) ----
        const uint16_t* prow = &pbuf[col * 32];
        v16bf ap = cat16(ld8(prow + half * 16), ld8(prow + half * 16 + 8));

        // ---- O += P V ----
#pragma unroll
        for (int t = 0; t < 4; ++t)
            o[t] = __builtin_amdgcn_wmma_f32_16x16x32_bf16(false, ap, false, bv[t], (short)0, o[t], false, false);
    }

    // ---- epilogue: O /= l, store bf16 [b, n, h*64 + d] ----
#pragma unroll
    for (int r = 0; r < 8; ++r) {
        const float inv = 1.0f / lrow[r];
        const int M = half * 8 + r;
        const size_t orow = (size_t)(b * SEQ + q0 + M) * DIM + h * HEADDIM;
#pragma unroll
        for (int t = 0; t < 4; ++t)
            attnb[orow + t * 16 + col] = f2bfu(o[t][r] * inv);
    }
}

// ---------------------------------------------------------------------------
// K3: out = attn @ w_o^T + b_o   -> fp32  [4096 x 1024]   (4096 waves)
// ---------------------------------------------------------------------------
__global__ __launch_bounds__(256)
void out_gemm_kernel(const uint16_t* __restrict__ ab,
                     const uint16_t* __restrict__ wb,
                     const float* __restrict__ bo,
                     float* __restrict__ out) {
    const int tid  = blockIdx.x * blockDim.x + threadIdx.x;
    const int wid  = tid >> 5;
    const int lane = tid & 31;
    const int ntiles = DIM / 64;                    // 16
    const int m0 = (wid / ntiles) * 16;
    const int n0 = (wid % ntiles) * 64;
    const int half = lane >> 4, col = lane & 15;

    v8f c[4] = {v8f{}, v8f{}, v8f{}, v8f{}};
    wave_gemm_16x64(ab + (size_t)(m0 + col) * DIM, wb, n0, DIM, lane, c);

#pragma unroll
    for (int t = 0; t < 4; ++t) {
        const int n = n0 + t * 16 + col;
        const float bias = bo[n];
#pragma unroll
        for (int r = 0; r < 8; ++r) {
            const int M = half * 8 + r;
            out[(size_t)(m0 + M) * DIM + n] = c[t][r] + bias;
        }
    }
}

// ---------------------------------------------------------------------------
// launch
// ---------------------------------------------------------------------------
extern "C" void kernel_launch(void* const* d_in, const int* in_sizes, int n_in,
                              void* d_out, int out_size, void* d_ws, size_t ws_size,
                              hipStream_t stream) {
    const float* x     = (const float*)d_in[0];   // [2,2048,1024]
    const float* w_qkv = (const float*)d_in[1];   // [3072,1024]
    const float* w_o   = (const float*)d_in[2];   // [1024,1024]
    const float* b_o   = (const float*)d_in[3];   // [1024]
    float* out = (float*)d_out;

    uint8_t* ws = (uint8_t*)d_ws;
    const size_t xb_elems    = (size_t)ROWS * DIM;            // 4,194,304
    const size_t wqkvb_elems = (size_t)QKVDIM * DIM;          // 3,145,728
    const size_t wob_elems   = (size_t)DIM * DIM;             // 1,048,576
    const size_t qkb_elems   = (size_t)ROWS * (2 * DIM);      // 8,388,608
    const size_t vtb_elems   = (size_t)BATCH * DIM * SEQ;     // 4,194,304
    const size_t attnb_elems = (size_t)ROWS * DIM;            // 4,194,304

    uint16_t* xb    = (uint16_t*)ws;                         ws += xb_elems * 2;
    uint16_t* wqkvb = (uint16_t*)ws;                         ws += wqkvb_elems * 2;
    uint16_t* wob   = (uint16_t*)ws;                         ws += wob_elems * 2;
    uint16_t* qkb   = (uint16_t*)ws;                         ws += qkb_elems * 2;
    uint16_t* vtb   = (uint16_t*)ws;                         ws += vtb_elems * 2;
    uint16_t* attnb = (uint16_t*)ws;                         (void)attnb_elems;

    // K0: fp32 -> bf16
    cvt_bf16_kernel<<<(int)((xb_elems    + 255) / 256), 256, 0, stream>>>(x,     xb,    (int)xb_elems);
    cvt_bf16_kernel<<<(int)((wqkvb_elems + 255) / 256), 256, 0, stream>>>(w_qkv, wqkvb, (int)wqkvb_elems);
    cvt_bf16_kernel<<<(int)((wob_elems   + 255) / 256), 256, 0, stream>>>(w_o,   wob,   (int)wob_elems);

    // K1: qkv projection — 12288 waves
    qkv_gemm_kernel<<<12288 * 32 / 256, 256, 0, stream>>>(xb, wqkvb, qkb, vtb);

    // K2: flash attention — 4096 waves
    attention_kernel<<<4096 * 32 / 256, 256, 0, stream>>>(qkb, vtb, attnb);

    // K3: output projection + bias — 4096 waves
    out_gemm_kernel<<<4096 * 32 / 256, 256, 0, stream>>>(attnb, wob, b_o, out);
}
